// StarTransformer_7421703487725
// MI455X (gfx1250) — compile-verified, matrix operand order checked by reading, not run on hardware
//
#include <hip/hip_runtime.h>
#include <hip/hip_bf16.h>

// ---------------- problem constants ----------------
#define BATCH 4
#define LLEN  1024
#define HDIM  1024          // == ND
#define NHEAD 16
#define HEADD 64
#define LPAD  1056          // 1025 rounded up to multiple of 32 for MSA2 keys
#define NITER 4

typedef __attribute__((ext_vector_type(16))) __bf16 v16bf;
typedef __attribute__((ext_vector_type(8)))  float  v8f;

union FragBF { uint4 u[2]; v16bf v; };

__device__ __forceinline__ void load_frag(FragBF& f, const __bf16* p) {
  f.u[0] = *(const uint4*)(p);
  f.u[1] = *(const uint4*)(p + 16);
}

__device__ __forceinline__ float wave_sum32(float v) {
#pragma unroll
  for (int off = 16; off; off >>= 1) v += __shfl_xor(v, off, 32);
  return v;
}

// ---------------------------------------------------------------------------
// WMMA GEMM:  C[b][m][n] = sum_k A[b][m][k] * W[n][k] + bias[n]
//   A: bf16 row-major [batch][M][1024], W: bf16 row-major [1024][1024]
//   C: f32  row-major [batch][M][1024]
// Block = 256 thr = 8 waves (2 M-waves x 4 N-waves) -> tile 32M x 256N.
// Each wave: 16M x 64N via 4 accumulators; K-loop unrolled 2x32 with
// register double-buffering so global_load_b128 clauses overlap the wmmas.
// Fragment layout per ISA 7.12.2 (16-bit, wave32):
//   lane<16 : K elems {kk+0..7, kk+16..23}; lane>=16: +8.
// ---------------------------------------------------------------------------
__global__ __launch_bounds__(256)
void wmma_gemm_bf16(const __bf16* __restrict__ A,
                    const __bf16* __restrict__ W,
                    const float*  __restrict__ bias,
                    float* __restrict__ C,
                    long strideA, long strideC)
{
  const int K = 1024, N = 1024;
  const int lane = threadIdx.x & 31;
  const int wave = threadIdx.x >> 5;
  const int wm = wave >> 2;            // 0..1
  const int wn = wave & 3;             // 0..3
  const int m0 = blockIdx.y * 32 + wm * 16;
  const int n0 = blockIdx.x * 256 + wn * 64;

  const __bf16* Ab = A + (long)blockIdx.z * strideA;
  float*        Cb = C + (long)blockIdx.z * strideC;

  const int kh = (lane >> 4) * 8;              // lane-group K offset
  const __bf16* aptr = Ab + (long)(m0 + (lane & 15)) * K + kh;
  const __bf16* w0   = W  + (long)(n0 +  0 + (lane & 15)) * K + kh;
  const __bf16* w1   = W  + (long)(n0 + 16 + (lane & 15)) * K + kh;
  const __bf16* w2   = W  + (long)(n0 + 32 + (lane & 15)) * K + kh;
  const __bf16* w3   = W  + (long)(n0 + 48 + (lane & 15)) * K + kh;

  v8f acc0 = {}, acc1 = {}, acc2 = {}, acc3 = {};

  // ping-pong fragment sets
  FragBF aA, b0A, b1A, b2A, b3A;
  FragBF aB, b0B, b1B, b2B, b3B;

  load_frag(aA,  aptr);
  load_frag(b0A, w0);
  load_frag(b1A, w1);
  load_frag(b2A, w2);
  load_frag(b3A, w3);

#pragma unroll 1
  for (int kk = 0; kk < K; kk += 64) {
    // issue loads for the odd half while computing the even half
    load_frag(aB,  aptr + kk + 32);
    load_frag(b0B, w0 + kk + 32);
    load_frag(b1B, w1 + kk + 32);
    load_frag(b2B, w2 + kk + 32);
    load_frag(b3B, w3 + kk + 32);
    __builtin_prefetch(aptr + kk + 64, 0, 3);   // near prefetch, 1 iter ahead
    __builtin_prefetch(w0 + kk + 64, 0, 3);
    __builtin_prefetch(w2 + kk + 64, 0, 3);

    acc0 = __builtin_amdgcn_wmma_f32_16x16x32_bf16(false, aA.v, false, b0A.v, (short)0, acc0, false, false);
    acc1 = __builtin_amdgcn_wmma_f32_16x16x32_bf16(false, aA.v, false, b1A.v, (short)0, acc1, false, false);
    acc2 = __builtin_amdgcn_wmma_f32_16x16x32_bf16(false, aA.v, false, b2A.v, (short)0, acc2, false, false);
    acc3 = __builtin_amdgcn_wmma_f32_16x16x32_bf16(false, aA.v, false, b3A.v, (short)0, acc3, false, false);

    if (kk + 64 < K) {                 // issue loads for next even half
      load_frag(aA,  aptr + kk + 64);
      load_frag(b0A, w0 + kk + 64);
      load_frag(b1A, w1 + kk + 64);
      load_frag(b2A, w2 + kk + 64);
      load_frag(b3A, w3 + kk + 64);
    }

    acc0 = __builtin_amdgcn_wmma_f32_16x16x32_bf16(false, aB.v, false, b0B.v, (short)0, acc0, false, false);
    acc1 = __builtin_amdgcn_wmma_f32_16x16x32_bf16(false, aB.v, false, b1B.v, (short)0, acc1, false, false);
    acc2 = __builtin_amdgcn_wmma_f32_16x16x32_bf16(false, aB.v, false, b2B.v, (short)0, acc2, false, false);
    acc3 = __builtin_amdgcn_wmma_f32_16x16x32_bf16(false, aB.v, false, b3B.v, (short)0, acc3, false, false);
  }

  // D layout: VGPR j -> row m0+j (lanes 0-15) / m0+j+8 (lanes 16-31), col = n + lane%16
  const int ncol  = lane & 15;
  const int mbase = m0 + (lane >> 4) * 8;
#pragma unroll
  for (int j = 0; j < 8; ++j) {
    long row = mbase + j;
    Cb[row * (long)N + n0 +  0 + ncol] = acc0[j] + bias[n0 +  0 + ncol];
    Cb[row * (long)N + n0 + 16 + ncol] = acc1[j] + bias[n0 + 16 + ncol];
    Cb[row * (long)N + n0 + 32 + ncol] = acc2[j] + bias[n0 + 32 + ncol];
    Cb[row * (long)N + n0 + 48 + ncol] = acc3[j] + bias[n0 + 48 + ncol];
  }
}

// ---------------- elementwise / small kernels ----------------
__global__ __launch_bounds__(256)
void k_init(const float* __restrict__ data, float* __restrict__ nodes,
            __bf16* __restrict__ embsb, long n)
{
  long i = (long)blockIdx.x * 256 + threadIdx.x;
  if (i >= n) return;
  float v = data[i];
  nodes[i] = v;
  embsb[i] = (__bf16)v;
}

__global__ __launch_bounds__(256)
void k_relay_init(const float* __restrict__ data, float* __restrict__ relay)
{
  int i = blockIdx.x * 256 + threadIdx.x;     // B*H
  if (i >= BATCH * HDIM) return;
  int b = i >> 10, h = i & 1023;
  float s = 0.f;
  for (int l = 0; l < LLEN; ++l) s += data[((long)b * LLEN + l) * HDIM + h];
  relay[i] = s * (1.f / LLEN);
}

__global__ __launch_bounds__(256)
void k_zero_bf(__bf16* __restrict__ p, long n)
{
  long i = (long)blockIdx.x * 256 + threadIdx.x;
  if (i < n) p[i] = (__bf16)0.0f;
}

__global__ __launch_bounds__(256)
void k_convert_bf(const float* __restrict__ s, __bf16* __restrict__ d, long n)
{
  long i = (long)blockIdx.x * 256 + threadIdx.x;
  if (i < n) d[i] = (__bf16)s[i];
}

// LayerNorm over channel dim (contiguous, position-major); one wave per position
__global__ __launch_bounds__(256)
void k_layernorm(const float* __restrict__ nodes, const float* __restrict__ sc,
                 const float* __restrict__ bi, __bf16* __restrict__ xnb)
{
  int pos  = blockIdx.x * 8 + (threadIdx.x >> 5);   // B*L positions
  int lane = threadIdx.x & 31;
  const float* x = nodes + (long)pos * HDIM;
  float sum = 0.f, sq = 0.f;
  for (int c = lane; c < HDIM; c += 32) { float v = x[c]; sum += v; sq += v * v; }
  sum = wave_sum32(sum);  sq = wave_sum32(sq);
  float mu   = sum * (1.f / HDIM);
  float var  = sq * (1.f / HDIM) - mu * mu;
  float rstd = rsqrtf(var + 1e-5f);
  __bf16* o = xnb + (long)pos * HDIM;
  for (int c = lane; c < HDIM; c += 32)
    o[c] = (__bf16)((x[c] - mu) * rstd * sc[c] + bi[c]);
}

// generic [B,1024] = W[1024x1024] @ x[B,1024] + bias  (tiny, f32)
__global__ __launch_bounds__(256)
void k_proj_bh(const float* __restrict__ W, const float* __restrict__ bias,
               const float* __restrict__ x, float* __restrict__ out, int lrelu)
{
  int i = blockIdx.x * 256 + threadIdx.x;     // B*1024
  if (i >= BATCH * HDIM) return;
  int b = i >> 10, o = i & 1023;
  const float* wr = W + (long)o * HDIM;
  const float* xv = x + (long)b * HDIM;
  float s = bias[o];
#pragma unroll 8
  for (int c = 0; c < HDIM; ++c) s += wr[c] * xv[c];
  if (lrelu) s = (s > 0.f) ? s : 0.01f * s;
  out[i] = s;
}

// MSA1 ring+star attention: one wave per (b,l,nh); 7 keys of length 64
__global__ __launch_bounds__(256)
void k_msa1_attn(const float* __restrict__ Q, const float* __restrict__ Kf,
                 const float* __restrict__ Vf, const float* __restrict__ AK,
                 const float* __restrict__ AV, const float* __restrict__ akr,
                 const float* __restrict__ avr, __bf16* __restrict__ attb)
{
  int w    = (blockIdx.x * 256 + threadIdx.x) >> 5;  // global wave id: B*L*NH
  int lane = threadIdx.x & 31;
  int nh = w & (NHEAD - 1);
  int l  = (w >> 4) & (LLEN - 1);
  int b  = w >> 14;
  long base = ((long)(b * LLEN + l)) * HDIM + nh * HEADD;
  int d0 = lane, d1 = lane + 32;
  float q0 = Q[base + d0], q1 = Q[base + d1];

  float sc[7];
#pragma unroll
  for (int u = 0; u < 5; ++u) {
    int lp = l - 2 + u;
    float p = 0.f;
    if (lp >= 0 && lp < LLEN) {
      long kb = ((long)(b * LLEN + lp)) * HDIM + nh * HEADD;
      p = q0 * Kf[kb + d0] + q1 * Kf[kb + d1];
    }
    sc[u] = wave_sum32(p);
  }
  sc[5] = wave_sum32(q0 * AK[base + d0] + q1 * AK[base + d1]);
  long rb = (long)b * HDIM + nh * HEADD;
  sc[6] = wave_sum32(q0 * akr[rb + d0] + q1 * akr[rb + d1]);

  float mx = -3.0e38f;
#pragma unroll
  for (int u = 0; u < 7; ++u) { sc[u] *= 0.125f; mx = fmaxf(mx, sc[u]); }
  float den = 0.f, al[7];
#pragma unroll
  for (int u = 0; u < 7; ++u) { al[u] = __expf(sc[u] - mx); den += al[u]; }
  float inv = 1.f / den;

  float a0 = 0.f, a1 = 0.f;
#pragma unroll
  for (int u = 0; u < 5; ++u) {
    int lp = l - 2 + u;
    if (lp >= 0 && lp < LLEN) {
      long vb = ((long)(b * LLEN + lp)) * HDIM + nh * HEADD;
      float al_u = al[u] * inv;
      a0 += al_u * Vf[vb + d0];
      a1 += al_u * Vf[vb + d1];
    }
  }
  a0 += al[5] * inv * AV[base + d0] + al[6] * inv * avr[rb + d0];
  a1 += al[5] * inv * AV[base + d1] + al[6] * inv * avr[rb + d1];
  attb[base + d0] = (__bf16)a0;
  attb[base + d1] = (__bf16)a1;
}

// nodes += leaky_relu(ring); write y rows 1..L (bf16)
__global__ __launch_bounds__(256)
void k_update_nodes(const float* __restrict__ ring, float* __restrict__ nodes,
                    __bf16* __restrict__ yb, long n)
{
  long i = (long)blockIdx.x * 256 + threadIdx.x;
  if (i >= n) return;
  int  h  = (int)(i & 1023);
  long bl = i >> 10;               // b*L + l
  int  b  = (int)(bl >> 10);
  int  l  = (int)(bl & 1023);
  float r = ring[i];
  float v = nodes[i] + ((r > 0.f) ? r : 0.01f * r);
  nodes[i] = v;
  yb[((long)(b * LPAD) + 1 + l) * HDIM + h] = (__bf16)v;
}

__global__ __launch_bounds__(256)
void k_y_row0(const float* __restrict__ relay, __bf16* __restrict__ yb)
{
  int i = blockIdx.x * 256 + threadIdx.x;   // B*H
  if (i >= BATCH * HDIM) return;
  int b = i >> 10, h = i & 1023;
  yb[(long)b * LPAD * HDIM + h] = (__bf16)relay[i];
}

// MSA2 scores+softmax over 1025 keys; one block per (b,nh)
__global__ __launch_bounds__(256)
void k_msa2_scores(const float* __restrict__ q2, const float* __restrict__ k2,
                   const int* __restrict__ mask, float* __restrict__ probs)
{
  int bnh = blockIdx.x;           // 0..63
  int b = bnh >> 4, nh = bnh & 15;
  int tid = threadIdx.x;
  __shared__ float red[256];
  const float* qv = q2 + (long)b * HDIM + nh * HEADD;
  float* pr = probs + (long)bnh * LPAD;

  float lmax = -3.0e38f;
  for (int l = tid; l < LLEN + 1; l += 256) {
    const float* kv = k2 + ((long)(b * LPAD + l)) * HDIM + nh * HEADD;
    float s = 0.f;
#pragma unroll 16
    for (int d = 0; d < HEADD; ++d) s += qv[d] * kv[d];
    s *= 0.125f;
    if (l > 0 && mask[b * LLEN + (l - 1)] == 0) s = -1e10f;
    pr[l] = s;
    lmax = fmaxf(lmax, s);
  }
  red[tid] = lmax; __syncthreads();
  for (int s = 128; s; s >>= 1) { if (tid < s) red[tid] = fmaxf(red[tid], red[tid + s]); __syncthreads(); }
  float mx = red[0]; __syncthreads();

  float lsum = 0.f;
  for (int l = tid; l < LLEN + 1; l += 256) {
    float e = __expf(pr[l] - mx);
    pr[l] = e; lsum += e;
  }
  red[tid] = lsum; __syncthreads();
  for (int s = 128; s; s >>= 1) { if (tid < s) red[tid] += red[tid + s]; __syncthreads(); }
  float inv = 1.f / red[0];
  for (int l = tid; l < LLEN + 1; l += 256) pr[l] *= inv;
}

// att2[b, o] = sum_l probs[b,nh,l] * k2[b,l,o]   (v == k per reference)
__global__ __launch_bounds__(256)
void k_msa2_att(const float* __restrict__ probs, const float* __restrict__ k2,
                float* __restrict__ att2)
{
  int i = blockIdx.x * 256 + threadIdx.x;   // B*ND
  if (i >= BATCH * HDIM) return;
  int b = i >> 10, o = i & 1023, nh = o >> 6;
  const float* pr = probs + (long)(b * NHEAD + nh) * LPAD;
  float s = 0.f;
  for (int l = 0; l < LLEN + 1; ++l)
    s += pr[l] * k2[((long)(b * LPAD + l)) * HDIM + o];
  att2[i] = s;
}

__global__ __launch_bounds__(256)
void k_mask_nodes(const int* __restrict__ mask, float* __restrict__ nodes, long n)
{
  long i = (long)blockIdx.x * 256 + threadIdx.x;
  if (i >= n) return;
  long bl = i >> 10;
  if (mask[bl] == 0) nodes[i] = 0.f;
}

__global__ __launch_bounds__(256)
void k_copy_out(const float* __restrict__ nodes, const float* __restrict__ relay,
                float* __restrict__ out, long nNodes, long nRelay)
{
  long i = (long)blockIdx.x * 256 + threadIdx.x;
  if (i < nNodes)                 out[i] = nodes[i];
  else if (i < nNodes + nRelay)   out[i] = relay[i - nNodes];
}

// ---------------------------------------------------------------------------
extern "C" void kernel_launch(void* const* d_in, const int* in_sizes, int n_in,
                              void* d_out, int out_size, void* d_ws, size_t ws_size,
                              hipStream_t stream)
{
  const float* data  = (const float*)d_in[0];
  const int*   mask  = (const int*)  d_in[1];
  const float* ln_s  = (const float*)d_in[2];
  const float* ln_b  = (const float*)d_in[3];
  const float* rwq   = (const float*)d_in[4];
  const float* rwq_b = (const float*)d_in[5];
  const float* rwk   = (const float*)d_in[6];
  const float* rwk_b = (const float*)d_in[7];
  const float* rwv   = (const float*)d_in[8];
  const float* rwv_b = (const float*)d_in[9];
  const float* rwo   = (const float*)d_in[10];
  const float* rwo_b = (const float*)d_in[11];
  const float* swq   = (const float*)d_in[12];
  const float* swq_b = (const float*)d_in[13];
  const float* swk   = (const float*)d_in[14];
  const float* swk_b = (const float*)d_in[15];
  const float* swo   = (const float*)d_in[16];
  const float* swo_b = (const float*)d_in[17];
  float* out = (float*)d_out;

  constexpr long   BLH = (long)BATCH * LLEN * HDIM;      // 4,194,304
  constexpr long   LPH = (long)BATCH * LPAD * HDIM;      // 4,325,376
  constexpr size_t WSZ = (size_t)HDIM * HDIM;            // 1,048,576

  char* ws = (char*)d_ws;
  size_t off = 0;
  auto take = [&](size_t bytes) { size_t o = off; off = (off + bytes + 255) & ~(size_t)255; return o; };

  float*  nodes = (float*) (ws + take(BLH * 4));
  __bf16* embsb = (__bf16*)(ws + take(BLH * 2));
  __bf16* xnb   = (__bf16*)(ws + take(BLH * 2));
  float*  Qf    = (float*) (ws + take(BLH * 4));   // also reused as ring output
  float*  Kf    = (float*) (ws + take(BLH * 4));
  float*  Vf    = (float*) (ws + take(BLH * 4));
  float*  AKf   = (float*) (ws + take(BLH * 4));
  float*  AVf   = (float*) (ws + take(BLH * 4));
  __bf16* attb  = (__bf16*)(ws + take(BLH * 2));
  __bf16* yb    = (__bf16*)(ws + take(LPH * 2));
  float*  k2f   = (float*) (ws + take(LPH * 4));
  __bf16* wq_s  = (__bf16*)(ws + take(WSZ * 2));
  __bf16* wk_s  = (__bf16*)(ws + take(WSZ * 2));
  __bf16* wv_s  = (__bf16*)(ws + take(WSZ * 2));
  __bf16* wo_s  = (__bf16*)(ws + take(WSZ * 2));
  __bf16* swk_s = (__bf16*)(ws + take(WSZ * 2));
  float*  akr   = (float*) (ws + take((size_t)BATCH * HDIM * 4));
  float*  avr   = (float*) (ws + take((size_t)BATCH * HDIM * 4));
  float*  q2    = (float*) (ws + take((size_t)BATCH * HDIM * 4));
  float*  att2  = (float*) (ws + take((size_t)BATCH * HDIM * 4));
  float*  relay = (float*) (ws + take((size_t)BATCH * HDIM * 4));
  float*  probs = (float*) (ws + take((size_t)BATCH * NHEAD * LPAD * 4));
  (void)in_sizes; (void)n_in; (void)out_size; (void)ws_size;

  const long SA = (long)LLEN * HDIM;      // per-batch activation stride
  const long SK = (long)LPAD * HDIM;      // per-batch padded-key stride

  // ---- init ----
  k_init      <<<(int)((BLH + 255) / 256), 256, 0, stream>>>(data, nodes, embsb, BLH);
  k_relay_init<<<16, 256, 0, stream>>>(data, relay);
  k_zero_bf   <<<(int)((LPH + 255) / 256), 256, 0, stream>>>(yb, LPH);

  dim3 gmain(4, 32, BATCH);   // N/256, M/32, batch  (M = L = 1024)
  dim3 gk2  (4, 33, BATCH);   // M = LPAD = 1056

  for (int it = 0; it < NITER; ++it) {
    const size_t wo_ = (size_t)it * WSZ;

    // per-iteration bf16 weight mirrors
    k_convert_bf<<<4096, 256, 0, stream>>>(rwq + wo_, wq_s,  (long)WSZ);
    k_convert_bf<<<4096, 256, 0, stream>>>(rwk + wo_, wk_s,  (long)WSZ);
    k_convert_bf<<<4096, 256, 0, stream>>>(rwv + wo_, wv_s,  (long)WSZ);
    k_convert_bf<<<4096, 256, 0, stream>>>(rwo + wo_, wo_s,  (long)WSZ);
    k_convert_bf<<<4096, 256, 0, stream>>>(swk + wo_, swk_s, (long)WSZ);

    // xn = LN(nodes) -> bf16
    k_layernorm<<<(BATCH * LLEN) / 8, 256, 0, stream>>>(nodes, ln_s + it * HDIM, ln_b + it * HDIM, xnb);

    // Q, K, V projections (WMMA)
    wmma_gemm_bf16<<<gmain, 256, 0, stream>>>(xnb, wq_s, rwq_b + it * HDIM, Qf, SA, SA);
    wmma_gemm_bf16<<<gmain, 256, 0, stream>>>(xnb, wk_s, rwk_b + it * HDIM, Kf, SA, SA);
    wmma_gemm_bf16<<<gmain, 256, 0, stream>>>(xnb, wv_s, rwv_b + it * HDIM, Vf, SA, SA);
    // AK/AV: projections of the (constant) embeddings
    wmma_gemm_bf16<<<gmain, 256, 0, stream>>>(embsb, wk_s, rwk_b + it * HDIM, AKf, SA, SA);
    wmma_gemm_bf16<<<gmain, 256, 0, stream>>>(embsb, wv_s, rwv_b + it * HDIM, AVf, SA, SA);
    // relay-column key/value (tiny)
    k_proj_bh<<<16, 256, 0, stream>>>(rwk + wo_, rwk_b + it * HDIM, relay, akr, 0);
    k_proj_bh<<<16, 256, 0, stream>>>(rwv + wo_, rwv_b + it * HDIM, relay, avr, 0);

    // ring attention -> bf16 att
    k_msa1_attn<<<(BATCH * LLEN * NHEAD) / 8, 256, 0, stream>>>(Qf, Kf, Vf, AKf, AVf, akr, avr, attb);

    // output projection (reuse Qf as ring buffer)
    wmma_gemm_bf16<<<gmain, 256, 0, stream>>>(attb, wo_s, rwo_b + it * HDIM, Qf, SA, SA);

    // nodes update + build y (relay row 0, nodes rows 1..L)
    k_y_row0      <<<16, 256, 0, stream>>>(relay, yb);
    k_update_nodes<<<(int)((BLH + 255) / 256), 256, 0, stream>>>(Qf, nodes, yb, BLH);

    // MSA2: keys (WMMA over padded 1056 rows), q, softmax, att, relay update
    wmma_gemm_bf16<<<gk2, 256, 0, stream>>>(yb, swk_s, swk_b + it * HDIM, k2f, SK, SK);
    k_proj_bh     <<<16, 256, 0, stream>>>(swq + wo_, swq_b + it * HDIM, relay, q2, 0);
    k_msa2_scores <<<BATCH * NHEAD, 256, 0, stream>>>(q2, k2f, mask, probs);
    k_msa2_att    <<<16, 256, 0, stream>>>(probs, k2f, att2);
    k_proj_bh     <<<16, 256, 0, stream>>>(swo + wo_, swo_b + it * HDIM, att2, relay, 1);

    // zero padded node positions (after relay update, per reference order)
    k_mask_nodes<<<(int)((BLH + 255) / 256), 256, 0, stream>>>(mask, nodes, BLH);
  }

  const long nOut = BLH + (long)BATCH * HDIM;
  k_copy_out<<<(int)((nOut + 255) / 256), 256, 0, stream>>>(nodes, relay, out, BLH, (long)BATCH * HDIM);
}
